// MS_SSA_Conv_21947282882701
// MI455X (gfx1250) — matrix operand order repaired and checked by reading, not compiled, not run on hardware
//
#include <hip/hip_runtime.h>
#include <hip/hip_bf16.h>

typedef __attribute__((ext_vector_type(16))) _Float16 v16h;
typedef __attribute__((ext_vector_type(8)))  _Float16 v8h;
typedef __attribute__((ext_vector_type(8)))  float    v8f;
typedef int i4v __attribute__((vector_size(16)));   // matches builtin param type

#define TT  4
#define BB  32
#define CC  384
#define NSP 256                       // H*W
#define SPIKE_ELEMS (TT*BB*CC*NSP)    // 12,582,912

// ---------------------------------------------------------------------------
// WMMA helper: D = A(16x32 f16) * B(32x16 f16) + C(16x16 f32)
// ---------------------------------------------------------------------------
__device__ __forceinline__ v8f wmma16x16x32(v16h a, v16h b, v8f c) {
#if defined(__HIP_DEVICE_COMPILE__)
    return __builtin_amdgcn_wmma_f32_16x16x32_f16(
        /*neg_a=*/false, a, /*neg_b=*/false, b,
        /*c_mod=*/(short)0, c, /*reuse_a=*/false, /*reuse_b=*/false);
#else
    (void)a; (void)b;
    return c;
#endif
}

// A-matrix fragment, 16x32 f16, row-major source, leading dim `ld` (elems).
// ISA 7.12.2: lane<16 -> row=lane, K {0..7,16..23}; lane>=16 -> row=lane-16, K {8..15,24..31}
__device__ __forceinline__ v16h ldA(const _Float16* __restrict__ base, int ld) {
    const int lane = threadIdx.x & 31;
    const int row  = lane & 15;
    const int kh   = (lane >> 4) << 3;          // 0 or 8
    const _Float16* p = base + (size_t)row * ld + kh;
    v8h lo = *(const v8h*)(p);
    v8h hi = *(const v8h*)(p + 16);
    return __builtin_shufflevector(lo, hi, 0,1,2,3,4,5,6,7,8,9,10,11,12,13,14,15);
}

// B-matrix fragment, 32x16 f16, source = weights row-major (Cout, Cin):
// lane<16 -> col=lane, K {0..7,16..23}; lane>=16 -> col=lane-16, K {8..15,24..31}
__device__ __forceinline__ v16h ldB(const _Float16* __restrict__ base, int ld) {
    const int lane = threadIdx.x & 31;
    const int col  = lane & 15;
    const int kb   = (lane >> 4) << 3;          // 0 or 8
    const _Float16* p = base + (size_t)col * ld + kb;
    v8h lo = *(const v8h*)(p);
    v8h hi = *(const v8h*)(p + 16);
    return __builtin_shufflevector(lo, hi, 0,1,2,3,4,5,6,7,8,9,10,11,12,13,14,15);
}

// ---------------------------------------------------------------------------
// CDNA5 async global->LDS copy (16B) + asynccnt waits.
// Builtin signature (from hipcc diagnostic): (int4 AS1*, int4 AS3*, Ii, Ii).
// ---------------------------------------------------------------------------
__device__ __forceinline__ void async_cp16(const void* g, void* l) {
#if defined(__HIP_DEVICE_COMPILE__)
#if __has_builtin(__builtin_amdgcn_global_load_async_to_lds_b128)
    __builtin_amdgcn_global_load_async_to_lds_b128(
        (__attribute__((address_space(1))) i4v*)g,
        (__attribute__((address_space(3))) i4v*)l, 0, 0);
#else
    unsigned loff = (unsigned)(size_t)(__attribute__((address_space(3))) void*)l;
    unsigned long long ga = (unsigned long long)(size_t)g;
    asm volatile("global_load_async_to_lds_b128 %0, %1, off" :: "v"(loff), "v"(ga) : "memory");
#endif
#else
    (void)g; (void)l;
#endif
}

__device__ __forceinline__ void wait_async_le8() {
#if defined(__HIP_DEVICE_COMPILE__)
#if __has_builtin(__builtin_amdgcn_s_wait_asynccnt)
    __builtin_amdgcn_s_wait_asynccnt(8);
#else
    asm volatile("s_wait_asynccnt 0x8" ::: "memory");
#endif
#endif
}

__device__ __forceinline__ void wait_async_le0() {
#if defined(__HIP_DEVICE_COMPILE__)
#if __has_builtin(__builtin_amdgcn_s_wait_asynccnt)
    __builtin_amdgcn_s_wait_asynccnt(0);
#else
    asm volatile("s_wait_asynccnt 0x0" ::: "memory");
#endif
#endif
}

// ---------------------------------------------------------------------------
// Kernel 1: convert the four weight matrices fp32 -> f16 (contiguous in ws)
// ---------------------------------------------------------------------------
__global__ void wcvt_kernel(const float* __restrict__ s0, const float* __restrict__ s1,
                            const float* __restrict__ s2, const float* __restrict__ s3,
                            _Float16* __restrict__ dst) {
    int i = blockIdx.x * blockDim.x + threadIdx.x;      // 0 .. 4*CC*CC
    const int WN = CC * CC;
    int which = i / WN;
    int r = i - which * WN;
    const float* s = (which == 0) ? s0 : (which == 1) ? s1 : (which == 2) ? s2 : s3;
    dst[i] = (_Float16)s[r];
}

// ---------------------------------------------------------------------------
// Kernel 2: shortcut LIF on x (T,B,C,N) -> binary spikes xs f16 in (t,b,n,c)
// layout. 32x32 LDS tile transpose: coalesced loads along n, coalesced f16
// stores along c. LIF state across t kept in registers (thread owns (c,n)).
// ---------------------------------------------------------------------------
__global__ void __launch_bounds__(256) lif_input_kernel(const float* __restrict__ x,
                                                        _Float16* __restrict__ xs) {
    __shared__ _Float16 tile[32][33];
    const int blk = blockIdx.x;          // B * 12 * 8 blocks
    const int b   = blk / 96;
    const int r   = blk - b * 96;
    const int c0  = (r >> 3) * 32;
    const int n0  = (r & 7) * 32;
    const int tid = threadIdx.x;
    const int ln  = tid & 31;
    const int g4  = tid >> 5;            // 0..7 (4 elements each)

    float vst[4] = {0.f, 0.f, 0.f, 0.f};
    for (int t = 0; t < TT; ++t) {
        #pragma unroll
        for (int j = 0; j < 4; ++j) {    // load along n (coalesced), LIF
            int c = c0 + g4 * 4 + j;
            float xv = x[((size_t)(t * BB + b) * CC + c) * NSP + n0 + ln];
            float vv = vst[j];
            vv += (xv - vv) * 0.5f;
            float s = (vv >= 1.0f) ? 1.0f : 0.0f;
            vst[j] = vv * (1.0f - s);
            tile[g4 * 4 + j][ln] = (_Float16)s;      // tile[c_local][n_local]
        }
        __syncthreads();
        #pragma unroll
        for (int j = 0; j < 4; ++j) {    // store along c (coalesced)
            int nl = g4 * 4 + j;
            xs[((size_t)(t * BB + b) * NSP + n0 + nl) * CC + c0 + ln] = tile[ln][nl];
        }
        __syncthreads();
    }
}

// ---------------------------------------------------------------------------
// Kernel 3: q/k/v branch: GEMM (spikes x W^T) -> BN -> LIF over t (in regs)
// Grid: (M/128, 384/64, 3 branches). Block 256 = 8 waves, wave tile 32x32.
// ---------------------------------------------------------------------------
__global__ void __launch_bounds__(256) qkv_gemm_bn_lif_kernel(
    const _Float16* __restrict__ xs,
    const _Float16* __restrict__ wq, const _Float16* __restrict__ wk, const _Float16* __restrict__ wv,
    const float* __restrict__ qg, const float* __restrict__ qbe, const float* __restrict__ qmu, const float* __restrict__ qva,
    const float* __restrict__ kg, const float* __restrict__ kbe, const float* __restrict__ kmu, const float* __restrict__ kva,
    const float* __restrict__ vg, const float* __restrict__ vbe, const float* __restrict__ vmu, const float* __restrict__ vva,
    _Float16* __restrict__ qs, _Float16* __restrict__ ks, _Float16* __restrict__ vs,
    float* __restrict__ v_out)
{
    const int lane = threadIdx.x & 31;
    const int wid  = threadIdx.x >> 5;
    const int wm   = wid & 3;
    const int wn   = wid >> 2;
    const int z    = blockIdx.z;

    const _Float16* W = (z == 0) ? wq : (z == 1) ? wk : wv;
    const float* g  = (z == 0) ? qg  : (z == 1) ? kg  : vg;
    const float* be = (z == 0) ? qbe : (z == 1) ? kbe : vbe;
    const float* mu = (z == 0) ? qmu : (z == 1) ? kmu : vmu;
    const float* va = (z == 0) ? qva : (z == 1) ? kva : vva;
    _Float16* outS  = (z == 0) ? qs  : (z == 1) ? ks  : vs;

    const int b_img = (blockIdx.x * 128) >> 8;
    const int row0  = (blockIdx.x * 128) & 255;
    const int wrow  = row0 + wm * 32;
    const int ncol0 = blockIdx.y * 64 + wn * 32;
    const int hi8   = (lane >> 4) << 3;
    const int cl    = lane & 15;

    __builtin_prefetch(W + (size_t)ncol0 * CC, 0, 3);

    float scale[2], shift[2];
    #pragma unroll
    for (int j = 0; j < 2; ++j) {
        int d = ncol0 + j * 16 + cl;
        float sc = g[d] * rsqrtf(va[d] + 1e-5f);
        scale[j] = sc;
        shift[j] = be[d] - mu[d] * sc;
    }

    v8f vst[2][2] = {};
    for (int t = 0; t < TT; ++t) {
        v8f acc[2][2] = {};
        const _Float16* Arow = xs + (size_t)((t * BB + b_img) * NSP + wrow) * CC;
        for (int k0 = 0; k0 < CC; k0 += 32) {
            v16h a0 = ldA(Arow + k0, CC);
            v16h a1 = ldA(Arow + (size_t)16 * CC + k0, CC);
            v16h b0 = ldB(W + (size_t)(ncol0     ) * CC + k0, CC);
            v16h b1 = ldB(W + (size_t)(ncol0 + 16) * CC + k0, CC);
            acc[0][0] = wmma16x16x32(a0, b0, acc[0][0]);
            acc[0][1] = wmma16x16x32(a0, b1, acc[0][1]);
            acc[1][0] = wmma16x16x32(a1, b0, acc[1][0]);
            acc[1][1] = wmma16x16x32(a1, b1, acc[1][1]);
        }
        #pragma unroll
        for (int i = 0; i < 2; ++i) {
            #pragma unroll
            for (int j = 0; j < 2; ++j) {
                int d = ncol0 + j * 16 + cl;
                #pragma unroll
                for (int e = 0; e < 8; ++e) {
                    float bnv = acc[i][j][e] * scale[j] + shift[j];
                    float vv  = vst[i][j][e];
                    vv += (bnv - vv) * 0.5f;
                    float s = (vv >= 1.0f) ? 1.0f : 0.0f;
                    vst[i][j][e] = vv * (1.0f - s);
                    int nsp = wrow + i * 16 + hi8 + e;
                    outS[((size_t)((t * BB + b_img) * NSP + nsp)) * CC + d] = (_Float16)s;
                    if (z == 2) {
                        int h = d / 48, ch = d - h * 48;
                        v_out[(((size_t)((t * BB + b_img) * 8 + h)) * NSP + nsp) * 48 + ch] = s;
                    }
                }
            }
        }
    }
}

// ---------------------------------------------------------------------------
// Kernel 4: kv[t,b,c] = sum_n k*v (binary AND-count), then LIF(v_th=0.5).
// One block per (b, 32-channel tile). Per t: async global->LDS staging of the
// 256x32 k and v spike tiles (double-buffered across t, ASYNC path),
// v8h packed products, 64-way LDS tree reduction, LIF in the npart==0 threads.
// ---------------------------------------------------------------------------
__global__ void __launch_bounds__(256) kv_reduce_lif_kernel(
    const _Float16* __restrict__ ksp, const _Float16* __restrict__ vsp,
    _Float16* __restrict__ kvs)
{
    __shared__ __align__(16) _Float16 sk[2][NSP * 32];   // 16 KB x2
    __shared__ __align__(16) _Float16 sv[2][NSP * 32];   // 16 KB x2
    __shared__ float red[64][33];                        // padded vs bank conflicts

    const int b   = blockIdx.x;
    const int c0  = blockIdx.y * 32;
    const int tid = threadIdx.x;
    const int cgrp  = tid & 3;       // 4 channel-groups of 8
    const int npart = tid >> 2;      // 64 partitions x 4 rows

    auto issue_copy = [&](int t, int buf) {
        const _Float16* gk = ksp + ((size_t)(t * BB + b) * NSP + tid) * CC + c0;
        const _Float16* gv = vsp + ((size_t)(t * BB + b) * NSP + tid) * CC + c0;
        _Float16* lk = &sk[buf][tid * 32];
        _Float16* lv = &sv[buf][tid * 32];
        #pragma unroll
        for (int i = 0; i < 4; ++i) {
            async_cp16(gk + i * 8, lk + i * 8);
            async_cp16(gv + i * 8, lv + i * 8);
        }
    };

    issue_copy(0, 0);
    float vstate[8] = {0.f, 0.f, 0.f, 0.f, 0.f, 0.f, 0.f, 0.f};

    for (int t = 0; t < TT; ++t) {
        const int buf = t & 1;
        if (t + 1 < TT) {
            issue_copy(t + 1, buf ^ 1);   // prefetch next timestep
            wait_async_le8();             // oldest batch (this t) complete
        } else {
            wait_async_le0();
        }
        __syncthreads();

        float part[8];
        #pragma unroll
        for (int i = 0; i < 8; ++i) part[i] = 0.f;
        #pragma unroll
        for (int r = 0; r < 4; ++r) {
            int n = npart * 4 + r;
            const v8h kk = *(const v8h*)&sk[buf][n * 32 + cgrp * 8];
            const v8h vv = *(const v8h*)&sv[buf][n * 32 + cgrp * 8];
            #pragma unroll
            for (int i = 0; i < 8; ++i) part[i] += (float)kk[i] * (float)vv[i];
        }
        #pragma unroll
        for (int i = 0; i < 8; ++i) red[npart][cgrp * 8 + i] = part[i];
        __syncthreads();
        #pragma unroll
        for (int s = 32; s >= 1; s >>= 1) {
            if (npart < s) {
                #pragma unroll
                for (int i = 0; i < 8; ++i)
                    red[npart][cgrp * 8 + i] += red[npart + s][cgrp * 8 + i];
            }
            __syncthreads();
        }
        if (npart == 0) {
            #pragma unroll
            for (int i = 0; i < 8; ++i) {
                float sum = red[0][cgrp * 8 + i];
                float vv = vstate[i];
                vv += (sum - vv) * 0.5f;
                float s = (vv >= 0.5f) ? 1.0f : 0.0f;
                vstate[i] = vv * (1.0f - s);
                kvs[(size_t)(t * BB + b) * CC + c0 + cgrp * 8 + i] = (_Float16)s;
            }
        }
        __syncthreads();
    }
}

// ---------------------------------------------------------------------------
// Kernel 5: y = (q * kv) @ proj_w^T + bias -> BN -> + identity -> out0
// ---------------------------------------------------------------------------
__global__ void __launch_bounds__(256) proj_gemm_kernel(
    const _Float16* __restrict__ qs, const _Float16* __restrict__ kvs,
    const _Float16* __restrict__ wp,
    const float* __restrict__ x,
    const float* __restrict__ pb, const float* __restrict__ pg, const float* __restrict__ pbe,
    const float* __restrict__ pmu, const float* __restrict__ pva,
    float* __restrict__ out0)
{
    const int lane = threadIdx.x & 31;
    const int wid  = threadIdx.x >> 5;
    const int wm   = wid & 3;
    const int wn   = wid >> 2;
    const int t    = blockIdx.z;

    const int b_img = (blockIdx.x * 128) >> 8;
    const int row0  = (blockIdx.x * 128) & 255;
    const int wrow  = row0 + wm * 32;
    const int ncol0 = blockIdx.y * 64 + wn * 32;
    const int hi8   = (lane >> 4) << 3;
    const int cl    = lane & 15;

    __builtin_prefetch(wp + (size_t)ncol0 * CC, 0, 3);

    float scale[2], shift[2];
    #pragma unroll
    for (int j = 0; j < 2; ++j) {
        int d = ncol0 + j * 16 + cl;
        float sc = pg[d] * rsqrtf(pva[d] + 1e-5f);
        scale[j] = sc;
        shift[j] = pbe[d] + (pb[d] - pmu[d]) * sc;
    }

    v8f acc[2][2] = {};
    const _Float16* Arow = qs  + (size_t)((t * BB + b_img) * NSP + wrow) * CC;
    const _Float16* KV   = kvs + (size_t)(t * BB + b_img) * CC;
    for (int k0 = 0; k0 < CC; k0 += 32) {
        v16h kf = ldA(KV + k0, 0);
        v16h a0 = ldA(Arow + k0, CC) * kf;
        v16h a1 = ldA(Arow + (size_t)16 * CC + k0, CC) * kf;
        v16h b0 = ldB(wp + (size_t)(ncol0     ) * CC + k0, CC);
        v16h b1 = ldB(wp + (size_t)(ncol0 + 16) * CC + k0, CC);
        acc[0][0] = wmma16x16x32(a0, b0, acc[0][0]);
        acc[0][1] = wmma16x16x32(a0, b1, acc[0][1]);
        acc[1][0] = wmma16x16x32(a1, b0, acc[1][0]);
        acc[1][1] = wmma16x16x32(a1, b1, acc[1][1]);
    }
    #pragma unroll
    for (int i = 0; i < 2; ++i) {
        #pragma unroll
        for (int j = 0; j < 2; ++j) {
            int d = ncol0 + j * 16 + cl;
            size_t base = ((size_t)(t * BB + b_img) * CC + d) * NSP + (size_t)(wrow + i * 16 + hi8);
            #pragma unroll
            for (int e = 0; e < 8; ++e) {
                float val = acc[i][j][e] * scale[j] + shift[j] + x[base + e];
                out0[base + e] = val;
            }
        }
    }
}

// ---------------------------------------------------------------------------
extern "C" void kernel_launch(void* const* d_in, const int* in_sizes, int n_in,
                              void* d_out, int out_size, void* d_ws, size_t ws_size,
                              hipStream_t stream) {
    (void)in_sizes; (void)n_in; (void)out_size; (void)ws_size;

    const float* x   = (const float*)d_in[0];
    const float* qw  = (const float*)d_in[1];
    const float* qg  = (const float*)d_in[2];
    const float* qbe = (const float*)d_in[3];
    const float* qmu = (const float*)d_in[4];
    const float* qva = (const float*)d_in[5];
    const float* kw  = (const float*)d_in[6];
    const float* kg  = (const float*)d_in[7];
    const float* kbe = (const float*)d_in[8];
    const float* kmu = (const float*)d_in[9];
    const float* kva = (const float*)d_in[10];
    const float* vw  = (const float*)d_in[11];
    const float* vg  = (const float*)d_in[12];
    const float* vbe = (const float*)d_in[13];
    const float* vmu = (const float*)d_in[14];
    const float* vva = (const float*)d_in[15];
    const float* pw  = (const float*)d_in[16];
    const float* pb  = (const float*)d_in[17];
    const float* pg  = (const float*)d_in[18];
    const float* pbe = (const float*)d_in[19];
    const float* pmu = (const float*)d_in[20];
    const float* pva = (const float*)d_in[21];

    float* out0  = (float*)d_out;
    float* v_out = out0 + (size_t)SPIKE_ELEMS;

    _Float16* xs   = (_Float16*)d_ws;
    _Float16* qs   = xs  + (size_t)SPIKE_ELEMS;
    _Float16* ks   = qs  + (size_t)SPIKE_ELEMS;
    _Float16* vs   = ks  + (size_t)SPIKE_ELEMS;
    _Float16* kvs  = vs  + (size_t)SPIKE_ELEMS;           // T*B*C
    _Float16* wq16 = kvs + (size_t)TT * BB * CC;
    _Float16* wk16 = wq16 + (size_t)CC * CC;
    _Float16* wv16 = wk16 + (size_t)CC * CC;
    _Float16* wp16 = wv16 + (size_t)CC * CC;

    wcvt_kernel<<<(4 * CC * CC) / 256, 256, 0, stream>>>(qw, kw, vw, pw, wq16);
    lif_input_kernel<<<BB * 12 * 8, 256, 0, stream>>>(x, xs);

    dim3 g3(64, 6, 3);
    qkv_gemm_bn_lif_kernel<<<g3, 256, 0, stream>>>(
        xs, wq16, wk16, wv16,
        qg, qbe, qmu, qva, kg, kbe, kmu, kva, vg, vbe, vmu, vva,
        qs, ks, vs, v_out);

    dim3 g4(BB, 12);
    kv_reduce_lif_kernel<<<g4, 256, 0, stream>>>(ks, vs, kvs);

    dim3 g5(64, 6, TT);
    proj_gemm_kernel<<<g5, 256, 0, stream>>>(
        qs, kvs, wp16, x, pb, pg, pbe, pmu, pva, out0);
}
